// CrossViewSwapAttention_63410897158940
// MI455X (gfx1250) — compile-verified, matrix operand order checked by reading, not run on hardware
//
#include <hip/hip_runtime.h>
#include <hip/hip_bf16.h>
#include <math.h>

typedef __bf16 bf16_t;
typedef __attribute__((ext_vector_type(16))) __bf16 v16bf;
typedef __attribute__((ext_vector_type(8)))  float  v8f;

#define DIMC   128
#define NVIEW  6
#define BEVN   128
#define EPSLN  1e-5f

// ---------------------------------------------------------------------------
// WMMA helpers (gfx1250 wave32, v_wmma_f32_16x16x32_bf16)
// ---------------------------------------------------------------------------
__device__ inline v8f wmma_bf(v16bf a, v16bf b, v8f c) {
  return __builtin_amdgcn_wmma_f32_16x16x32_bf16(false, a, false, b, (short)0, c,
                                                 false, false);
}

__device__ inline v8f vzero8() {
  v8f v;
#pragma unroll
  for (int i = 0; i < 8; ++i) v[i] = 0.0f;
  return v;
}

// 16x32 bf16 fragment load. Element (r,k) taken from p[r*sR + k*sK].
// For the A matrix r is the row (M); for B (stored as N x K) r is the column.
// ISA layout: lane L: r = L&15 ; k-half = (L>>4)*8 ; VGPR j holds k-pair
//   k = (j>>2)*16 + half + (j&3)*2.
__device__ inline v16bf ldfrag_bf(const bf16_t* p, int sR, int sK) {
  int l = threadIdx.x & 31;
  int r = l & 15;
  int kh = (l >> 4) * 8;
  v16bf v;
#pragma unroll
  for (int j = 0; j < 8; ++j) {
    int k = ((j >> 2) << 4) + kh + ((j & 3) << 1);
    v[2 * j]     = p[r * sR + k * sK];
    v[2 * j + 1] = p[r * sR + (k + 1) * sK];
  }
  return v;
}

__device__ inline v16bf ldfrag_f32(const float* p, int sR, int sK) {
  int l = threadIdx.x & 31;
  int r = l & 15;
  int kh = (l >> 4) * 8;
  v16bf v;
#pragma unroll
  for (int j = 0; j < 8; ++j) {
    int k = ((j >> 2) << 4) + kh + ((j & 3) << 1);
    v[2 * j]     = (bf16_t)p[r * sR + k * sK];
    v[2 * j + 1] = (bf16_t)p[r * sR + (k + 1) * sK];
  }
  return v;
}

// D/C element (i) coordinates for this lane: n = L&15, m = (L>>4)*8 + i
__device__ inline void dcoord(int i, int& m, int& n) {
  int l = threadIdx.x & 31;
  n = l & 15;
  m = ((l >> 4) << 3) + i;
}

// ---------------------------------------------------------------------------
// 1) camera-center embedding: c_embed[n][o] = cam_w[o][:] . E_inv[n][:,3]
// ---------------------------------------------------------------------------
__global__ void k_cam_embed(const float* __restrict__ cam_w,
                            const float* __restrict__ E_inv,
                            float* __restrict__ c_embed) {
  int nv = blockIdx.x, o = threadIdx.x;
  float s = 0.f;
#pragma unroll
  for (int c = 0; c < 4; ++c) s += cam_w[o * 4 + c] * E_inv[nv * 16 + c * 4 + 3];
  c_embed[nv * DIMC + o] = s;
}

// ---------------------------------------------------------------------------
// 2) image positional embedding (normalized), token-major imgT[(n*1024+p)][o]
// ---------------------------------------------------------------------------
__global__ void k_img_embed(const float* __restrict__ img_w,
                            const float* __restrict__ I_inv,
                            const float* __restrict__ E_inv,
                            const float* __restrict__ c_embed,
                            float* __restrict__ imgT) {
  int tok = blockIdx.x;
  int nv = tok >> 10, p = tok & 1023;
  int py = p >> 5, px = p & 31;
  float pix0 = (px * (1.0f / 31.0f)) * 256.0f;
  float pix1 = (py * (1.0f / 31.0f)) * 256.0f;
  const float* I = I_inv + nv * 9;
  float cam[4];
#pragma unroll
  for (int r = 0; r < 3; ++r)
    cam[r] = I[r * 3 + 0] * pix0 + I[r * 3 + 1] * pix1 + I[r * 3 + 2];
  cam[3] = 1.0f;
  const float* E = E_inv + nv * 16;
  float d[4];
#pragma unroll
  for (int r = 0; r < 4; ++r) {
    float s = 0.f;
#pragma unroll
    for (int c = 0; c < 4; ++c) s += E[r * 4 + c] * cam[c];
    d[r] = s;
  }
  int o = threadIdx.x;
  float e = img_w[o * 4 + 0] * d[0] + img_w[o * 4 + 1] * d[1] +
            img_w[o * 4 + 2] * d[2] + img_w[o * 4 + 3] * d[3] -
            c_embed[nv * DIMC + o];
  float s = e * e;
  for (int off = 16; off; off >>= 1) s += __shfl_xor(s, off, 32);
  __shared__ float part[4];
  if ((threadIdx.x & 31) == 0) part[threadIdx.x >> 5] = s;
  __syncthreads();
  float tot = part[0] + part[1] + part[2] + part[3];
  imgT[(size_t)tok * DIMC + o] = e / (sqrtf(tot) + 1e-7f);
}

// ---------------------------------------------------------------------------
// 3) key/val: keyT = imgT + relu(bn(feature)) @ fprojW^T ; valT similarly
//    WMMA GEMM over 16-token tiles.
// ---------------------------------------------------------------------------
__global__ void k_keyval(const float* __restrict__ feature,
                         const float* __restrict__ fproj_g, const float* __restrict__ fproj_b,
                         const float* __restrict__ fproj_w,
                         const float* __restrict__ flin_g, const float* __restrict__ flin_b,
                         const float* __restrict__ flin_w,
                         const float* __restrict__ imgT,
                         float* __restrict__ keyT, float* __restrict__ valT) {
  __shared__ bf16_t actK[16][DIMC];
  __shared__ bf16_t actV[16][DIMC];
  int t0 = blockIdx.x * 16;
  int c = threadIdx.x;
  float invs = rsqrtf(1.0f + EPSLN);
  float gk = fproj_g[c] * invs, bk = fproj_b[c];
  float gv = flin_g[c] * invs, bv = flin_b[c];
#pragma unroll 4
  for (int m = 0; m < 16; ++m) {
    int tok = t0 + m;
    int nv = tok >> 10, p = tok & 1023;
    float f = feature[((size_t)nv * DIMC + c) * 1024 + p];
    actK[m][c] = (bf16_t)fmaxf(f * gk + bk, 0.f);
    actV[m][c] = (bf16_t)fmaxf(f * gv + bv, 0.f);
  }
  __syncthreads();
  int wv = threadIdx.x >> 5;
  const float* W = (wv < 2) ? fproj_w : flin_w;
  const bf16_t(*act)[DIMC] = (wv < 2) ? actK : actV;
  float* dst = (wv < 2) ? keyT : valT;
  int nt0 = (wv & 1) * 4;
  for (int nt = nt0; nt < nt0 + 4; ++nt) {
    v8f acc = vzero8();
#pragma unroll
    for (int ks = 0; ks < 4; ++ks) {
      v16bf a = ldfrag_bf(&act[0][ks * 32], DIMC, 1);
      v16bf b = ldfrag_f32(&W[(size_t)(nt * 16) * DIMC + ks * 32], DIMC, 1);
      acc = wmma_bf(a, b, acc);
    }
#pragma unroll
    for (int i = 0; i < 8; ++i) {
      int m, n; dcoord(i, m, n); n += nt * 16;
      size_t o = (size_t)(t0 + m) * DIMC + n;
      float v = acc[i];
      if (wv < 2) v += imgT[o];
      dst[o] = v;
    }
  }
}

// ---------------------------------------------------------------------------
// 4) generic fused LayerNorm + Linear(128->128), bf16 output (QKV projections)
// ---------------------------------------------------------------------------
__global__ void k_ln_linear(const float* __restrict__ src,
                            const float* __restrict__ g, const float* __restrict__ bln,
                            const float* __restrict__ W, const float* __restrict__ bias,
                            bf16_t* __restrict__ dst) {
  __shared__ bf16_t xn[16][DIMC];
  int t0 = blockIdx.x * 16;
  int m = threadIdx.x >> 3, cg = (threadIdx.x & 7) * 16;
  const float* row = src + (size_t)(t0 + m) * DIMC;
  float v[16], s = 0.f, s2 = 0.f;
#pragma unroll
  for (int j = 0; j < 16; ++j) { v[j] = row[cg + j]; s += v[j]; s2 += v[j] * v[j]; }
  for (int off = 4; off; off >>= 1) { s += __shfl_xor(s, off, 32); s2 += __shfl_xor(s2, off, 32); }
  float mean = s * (1.0f / DIMC);
  float var = s2 * (1.0f / DIMC) - mean * mean;
  float inv = rsqrtf(var + EPSLN);
#pragma unroll
  for (int j = 0; j < 16; ++j)
    xn[m][cg + j] = (bf16_t)((v[j] - mean) * inv * g[cg + j] + bln[cg + j]);
  __syncthreads();
  int wv = threadIdx.x >> 5;
  for (int nt = wv * 2; nt < wv * 2 + 2; ++nt) {
    v8f acc = vzero8();
#pragma unroll
    for (int ks = 0; ks < 4; ++ks) {
      v16bf a = ldfrag_bf(&xn[0][ks * 32], DIMC, 1);
      v16bf b = ldfrag_f32(&W[(size_t)(nt * 16) * DIMC + ks * 32], DIMC, 1);
      acc = wmma_bf(a, b, acc);
    }
#pragma unroll
    for (int i = 0; i < 8; ++i) {
      int mm, nn; dcoord(i, mm, nn); nn += nt * 16;
      dst[(size_t)(t0 + mm) * DIMC + nn] = (bf16_t)(acc[i] + bias[nn]);
    }
  }
}

// ---------------------------------------------------------------------------
// 5) attn1 Q build: q = normalize(bev_embed - c_embed) + x ; LN ; @WqT -> bf16
//    token order: n*16384 + y*128 + x
// ---------------------------------------------------------------------------
__global__ void k_build_q1(const float* __restrict__ xin,
                           const float* __restrict__ bev_w, const float* __restrict__ bev_b,
                           const float* __restrict__ c_embed,
                           const float* __restrict__ g, const float* __restrict__ bln,
                           const float* __restrict__ W, const float* __restrict__ bias,
                           bf16_t* __restrict__ dst) {
  __shared__ bf16_t xn[16][DIMC];
  int t0 = blockIdx.x * 16;
  int m = threadIdx.x >> 3, cg = (threadIdx.x & 7) * 16;
  int tok = t0 + m;
  int nv = tok >> 14;
  int pos = tok & 16383;
  int y = pos >> 7, x = pos & 127;
  // world coords (V^-1 of the BEV grid transform)
  const float sh = 1.28f, sw = 1.28f, cc = 64.0f;
  float w0 = (cc - (y * (128.0f / 127.0f))) / sh;
  float w1 = (cc - (x * (128.0f / 127.0f))) / sw;
  float e[16], s2 = 0.f;
#pragma unroll
  for (int j = 0; j < 16; ++j) {
    int c = cg + j;
    e[j] = bev_w[c * 2] * w0 + bev_w[c * 2 + 1] * w1 + bev_b[c] -
           c_embed[nv * DIMC + c];
    s2 += e[j] * e[j];
  }
  for (int off = 4; off; off >>= 1) s2 += __shfl_xor(s2, off, 32);
  float inv_n = 1.0f / (sqrtf(s2) + 1e-7f);
  float q[16], s = 0.f, ss = 0.f;
#pragma unroll
  for (int j = 0; j < 16; ++j) {
    q[j] = e[j] * inv_n + xin[(size_t)(cg + j) * 16384 + pos];
    s += q[j]; ss += q[j] * q[j];
  }
  for (int off = 4; off; off >>= 1) { s += __shfl_xor(s, off, 32); ss += __shfl_xor(ss, off, 32); }
  float mean = s * (1.0f / DIMC);
  float var = ss * (1.0f / DIMC) - mean * mean;
  float inv = rsqrtf(var + EPSLN);
#pragma unroll
  for (int j = 0; j < 16; ++j)
    xn[m][cg + j] = (bf16_t)((q[j] - mean) * inv * g[cg + j] + bln[cg + j]);
  __syncthreads();
  int wv = threadIdx.x >> 5;
  for (int nt = wv * 2; nt < wv * 2 + 2; ++nt) {
    v8f acc = vzero8();
#pragma unroll
    for (int ks = 0; ks < 4; ++ks) {
      v16bf a = ldfrag_bf(&xn[0][ks * 32], DIMC, 1);
      v16bf b = ldfrag_f32(&W[(size_t)(nt * 16) * DIMC + ks * 32], DIMC, 1);
      acc = wmma_bf(a, b, acc);
    }
#pragma unroll
    for (int i = 0; i < 8; ++i) {
      int mm, nn; dcoord(i, mm, nn); nn += nt * 16;
      dst[(size_t)(t0 + mm) * DIMC + nn] = (bf16_t)(acc[i] + bias[nn]);
    }
  }
}

// ---------------------------------------------------------------------------
// 6) init q1 with skip1 = x (channel-last)
// ---------------------------------------------------------------------------
__global__ void k_init_q1(const float* __restrict__ xin, float* __restrict__ q1) {
  int pos = blockIdx.x, c = threadIdx.x;
  q1[(size_t)pos * DIMC + c] = xin[(size_t)c * 16384 + pos];
}

// ---------------------------------------------------------------------------
// 7) windowed cross attention. MODE 0: attn1 (per-view, mean via atomics onto
//    skip-initialized out). MODE 1: attn2 (broadcast query, direct store+skip).
//    block = 128 threads (4 waves), each wave does 4 M-tiles of 16 q rows.
// ---------------------------------------------------------------------------
template <int MODE>
__global__ void k_attn(const bf16_t* __restrict__ Qh, const bf16_t* __restrict__ Kh,
                       const bf16_t* __restrict__ Vh,
                       const float* __restrict__ Wp, const float* __restrict__ bp,
                       const float* __restrict__ skipsrc, float* __restrict__ out) {
  __shared__ bf16_t Kl[96][DIMC];
  __shared__ bf16_t Vl[96][DIMC];
  __shared__ bf16_t Pl[4][16][96];

  int iy = blockIdx.y >> 3, jx = blockIdx.y & 7;
  int nv = blockIdx.x;

  // stage K/V window into LDS (96 tokens x 128)
  for (int idx = threadIdx.x; idx < 96 * DIMC; idx += 128) {
    int r = idx >> 7, c = idx & 127;
    int kn = r >> 4, f1 = (r >> 2) & 3, f2 = r & 3;
    size_t tok;
    if (MODE == 0)
      tok = (size_t)kn * 1024 + (iy * 4 + f1) * 32 + (jx * 4 + f2);
    else  // swapped window + swapped inner order
      tok = (size_t)kn * 1024 + (jx * 4 + f2) * 32 + (iy * 4 + f1);
    Kl[r][c] = Kh[tok * DIMC + c];
    Vl[r][c] = Vh[tok * DIMC + c];
  }
  __syncthreads();

  int wv = threadIdx.x >> 5;
  const float SCL = 0.17677669529663687f;  // 32^-0.5

  for (int mt = 0; mt < 4; ++mt) {
    int r0 = wv * 64 + mt * 16;
    int wy = r0 >> 4;  // rows in this tile share wy; lane r = wx
    size_t qbase = ((size_t)(MODE == 0 ? nv * 16384 : 0) +
                    (size_t)(iy * 16 + wy) * 128 + jx * 16) * DIMC;
    const bf16_t* Qrow = Qh + qbase;

    v8f pacc[8];
#pragma unroll
    for (int t = 0; t < 8; ++t) pacc[t] = vzero8();

    for (int h = 0; h < 4; ++h) {
      // ---- scores: S(16x96) = Q(16x32) x K^T ----
      v16bf aq = ldfrag_bf(Qrow + h * 32, DIMC, 1);
      v8f sc[6];
#pragma unroll
      for (int nt = 0; nt < 6; ++nt) {
        v16bf b = ldfrag_bf(&Kl[nt * 16][h * 32], DIMC, 1);
        sc[nt] = wmma_bf(aq, b, vzero8());
      }
      // ---- softmax over 96 keys (rows live in 16-lane halves) ----
#pragma unroll
      for (int i = 0; i < 8; ++i) {
        float mx = -1e30f;
#pragma unroll
        for (int j = 0; j < 6; ++j) mx = fmaxf(mx, sc[j][i]);
        for (int off = 8; off; off >>= 1) mx = fmaxf(mx, __shfl_xor(mx, off, 32));
        float sum = 0.f, ev[6];
#pragma unroll
        for (int j = 0; j < 6; ++j) { ev[j] = __expf((sc[j][i] - mx) * SCL); sum += ev[j]; }
        for (int off = 8; off; off >>= 1) sum += __shfl_xor(sum, off, 32);
        float rinv = 1.0f / sum;
#pragma unroll
        for (int j = 0; j < 6; ++j) sc[j][i] = ev[j] * rinv;
      }
      // P -> LDS (per-wave buffer)
      {
        int l = threadIdx.x & 31, n = l & 15, mh = (l >> 4) << 3;
#pragma unroll
        for (int j = 0; j < 6; ++j)
#pragma unroll
          for (int i = 0; i < 8; ++i) Pl[wv][mh + i][j * 16 + n] = (bf16_t)sc[j][i];
      }
      __syncthreads();
      // ---- O(16x32) = P(16x96) x V(96x32 slice) ----
      v8f ov[2];
      ov[0] = vzero8(); ov[1] = vzero8();
#pragma unroll
      for (int ks = 0; ks < 3; ++ks) {
        v16bf ap = ldfrag_bf(&Pl[wv][0][ks * 32], 96, 1);
#pragma unroll
        for (int nt2 = 0; nt2 < 2; ++nt2) {
          v16bf bv = ldfrag_bf(&Vl[ks * 32][h * 32 + nt2 * 16], 1, DIMC);
          ov[nt2] = wmma_bf(ap, bv, ov[nt2]);
        }
      }
      __syncthreads();
      // O -> LDS (reuse P region, first 32 cols)
      {
        int l = threadIdx.x & 31, n = l & 15, mh = (l >> 4) << 3;
#pragma unroll
        for (int nt2 = 0; nt2 < 2; ++nt2)
#pragma unroll
          for (int i = 0; i < 8; ++i) Pl[wv][mh + i][nt2 * 16 + n] = (bf16_t)ov[nt2][i];
      }
      __syncthreads();
      // ---- fused projection partial: k-slice = this head ----
      v16bf ao = ldfrag_bf(&Pl[wv][0][0], 96, 1);
#pragma unroll
      for (int nt = 0; nt < 8; ++nt) {
        v16bf bw = ldfrag_f32(&Wp[(size_t)(nt * 16) * DIMC + h * 32], DIMC, 1);
        pacc[nt] = wmma_bf(ao, bw, pacc[nt]);
      }
      __syncthreads();
    }  // heads

    // ---- write: MODE0 mean over views (atomic), MODE1 direct + skip ----
#pragma unroll
    for (int nt = 0; nt < 8; ++nt) {
#pragma unroll
      for (int i = 0; i < 8; ++i) {
        int mm, nn; dcoord(i, mm, nn); nn += nt * 16;
        int y = iy * 16 + wy, xp = jx * 16 + mm;
        size_t o = ((size_t)y * 128 + xp) * DIMC + nn;
        float v = pacc[nt][i] + bp[nn];
        if (MODE == 0)
          atomicAdd(&out[o], v * (1.0f / 6.0f));
        else
          out[o] = v + skipsrc[o];
      }
    }
  }
}

// ---------------------------------------------------------------------------
// 8) residual MLP: dst = src + fc2(gelu(fc1(LN(src))))
// ---------------------------------------------------------------------------
__global__ void k_mlp(const float* __restrict__ src,
                      const float* __restrict__ g, const float* __restrict__ bln,
                      const float* __restrict__ W1, const float* __restrict__ b1,
                      const float* __restrict__ W2, const float* __restrict__ b2,
                      float* __restrict__ dst) {
  __shared__ bf16_t xn[16][DIMC];
  __shared__ bf16_t hb[16][2 * DIMC];
  int t0 = blockIdx.x * 16;
  int m = threadIdx.x >> 3, cg = (threadIdx.x & 7) * 16;
  const float* row = src + (size_t)(t0 + m) * DIMC;
  float v[16], s = 0.f, s2 = 0.f;
#pragma unroll
  for (int j = 0; j < 16; ++j) { v[j] = row[cg + j]; s += v[j]; s2 += v[j] * v[j]; }
  for (int off = 4; off; off >>= 1) { s += __shfl_xor(s, off, 32); s2 += __shfl_xor(s2, off, 32); }
  float mean = s * (1.0f / DIMC);
  float var = s2 * (1.0f / DIMC) - mean * mean;
  float inv = rsqrtf(var + EPSLN);
#pragma unroll
  for (int j = 0; j < 16; ++j)
    xn[m][cg + j] = (bf16_t)((v[j] - mean) * inv * g[cg + j] + bln[cg + j]);
  __syncthreads();
  int wv = threadIdx.x >> 5;
  for (int nt = wv * 4; nt < wv * 4 + 4; ++nt) {
    v8f acc = vzero8();
#pragma unroll
    for (int ks = 0; ks < 4; ++ks) {
      v16bf a = ldfrag_bf(&xn[0][ks * 32], DIMC, 1);
      v16bf b = ldfrag_f32(&W1[(size_t)(nt * 16) * DIMC + ks * 32], DIMC, 1);
      acc = wmma_bf(a, b, acc);
    }
#pragma unroll
    for (int i = 0; i < 8; ++i) {
      int mm, nn; dcoord(i, mm, nn); nn += nt * 16;
      float h = acc[i] + b1[nn];
      h = 0.5f * h * (1.0f + erff(h * 0.7071067811865476f));  // exact gelu
      hb[mm][nn] = (bf16_t)h;
    }
  }
  __syncthreads();
  for (int nt = wv * 2; nt < wv * 2 + 2; ++nt) {
    v8f acc = vzero8();
#pragma unroll
    for (int ks = 0; ks < 8; ++ks) {
      v16bf a = ldfrag_bf(&hb[0][ks * 32], 2 * DIMC, 1);
      v16bf b = ldfrag_f32(&W2[(size_t)(nt * 16) * (2 * DIMC) + ks * 32], 2 * DIMC, 1);
      acc = wmma_bf(a, b, acc);
    }
#pragma unroll
    for (int i = 0; i < 8; ++i) {
      int mm, nn; dcoord(i, mm, nn); nn += nt * 16;
      size_t o = (size_t)(t0 + mm) * DIMC + nn;
      dst[o] = src[o] + acc[i] + b2[nn];
    }
  }
}

// ---------------------------------------------------------------------------
// 9) bottleneck stage A: y1 = relu(bn1(src @ w1^T))  (128 -> 32)
// ---------------------------------------------------------------------------
__global__ void k_bneck_a(const float* __restrict__ src, const float* __restrict__ w1,
                          const float* __restrict__ g1, const float* __restrict__ b1,
                          float* __restrict__ y1) {
  __shared__ bf16_t xn[16][DIMC];
  int t0 = blockIdx.x * 16;
  for (int idx = threadIdx.x; idx < 16 * DIMC; idx += 64)
    xn[idx >> 7][idx & 127] = (bf16_t)src[(size_t)t0 * DIMC + idx];
  __syncthreads();
  int nt = threadIdx.x >> 5;  // 2 waves, 1 n-tile each
  v8f acc = vzero8();
#pragma unroll
  for (int ks = 0; ks < 4; ++ks) {
    v16bf a = ldfrag_bf(&xn[0][ks * 32], DIMC, 1);
    v16bf b = ldfrag_f32(&w1[(size_t)(nt * 16) * DIMC + ks * 32], DIMC, 1);
    acc = wmma_bf(a, b, acc);
  }
  float invs = rsqrtf(1.0f + EPSLN);
#pragma unroll
  for (int i = 0; i < 8; ++i) {
    int mm, nn; dcoord(i, mm, nn); nn += nt * 16;
    float v = acc[i] * g1[nn] * invs + b1[nn];
    y1[(size_t)(t0 + mm) * 32 + nn] = fmaxf(v, 0.f);
  }
}

// ---------------------------------------------------------------------------
// 10) bottleneck stage B: 3x3 conv (32->32) + bn + relu, VALU from LDS tile
// ---------------------------------------------------------------------------
__global__ void k_conv3(const float* __restrict__ y1, const float* __restrict__ w2,
                        const float* __restrict__ g2, const float* __restrict__ b2,
                        float* __restrict__ y2) {
  __shared__ float tile[18][18][32];
  int by = (blockIdx.x >> 3) * 16, bx = (blockIdx.x & 7) * 16;
  for (int idx = threadIdx.x; idx < 18 * 18 * 32; idx += 256) {
    int c = idx & 31, sp = idx >> 5;
    int ly = sp / 18, lx = sp % 18;
    int gy = by + ly - 1, gx = bx + lx - 1;
    float v = 0.f;
    if (gy >= 0 && gy < 128 && gx >= 0 && gx < 128)
      v = y1[((size_t)(gy << 7) + gx) * 32 + c];
    tile[ly][lx][c] = v;
  }
  __syncthreads();
  int py = threadIdx.x >> 4, px = threadIdx.x & 15;
  float invs = rsqrtf(1.0f + EPSLN);
  for (int oc = 0; oc < 32; ++oc) {
    float s = 0.f;
#pragma unroll
    for (int ky = 0; ky < 3; ++ky)
#pragma unroll
      for (int kx = 0; kx < 3; ++kx) {
        const float* wr = &w2[((size_t)(oc * 32) * 3 + ky) * 3 + kx];
        const float* tr = &tile[py + ky][px + kx][0];
#pragma unroll
        for (int ic = 0; ic < 32; ++ic) s += wr[(size_t)ic * 9] * tr[ic];
      }
    float v = s * g2[oc] * invs + b2[oc];
    y2[((size_t)((by + py) << 7) + (bx + px)) * 32 + oc] = fmaxf(v, 0.f);
  }
}

// ---------------------------------------------------------------------------
// 11) bottleneck stage C: dst = relu(bn3(y2 @ w3^T) + resid)  (32 -> 128)
// ---------------------------------------------------------------------------
__global__ void k_bneck_c(const float* __restrict__ y2, const float* __restrict__ w3,
                          const float* __restrict__ g3, const float* __restrict__ b3,
                          const float* __restrict__ resid, float* __restrict__ dst) {
  __shared__ bf16_t xn[16][32];
  int t0 = blockIdx.x * 16;
  for (int idx = threadIdx.x; idx < 16 * 32; idx += 128)
    xn[idx >> 5][idx & 31] = (bf16_t)y2[(size_t)t0 * 32 + idx];
  __syncthreads();
  int wv = threadIdx.x >> 5;
  float invs = rsqrtf(1.0f + EPSLN);
  for (int nt = wv * 2; nt < wv * 2 + 2; ++nt) {
    v8f acc = vzero8();
    v16bf a = ldfrag_bf(&xn[0][0], 32, 1);
    v16bf b = ldfrag_f32(&w3[(size_t)(nt * 16) * 32], 32, 1);
    acc = wmma_bf(a, b, acc);
#pragma unroll
    for (int i = 0; i < 8; ++i) {
      int mm, nn; dcoord(i, mm, nn); nn += nt * 16;
      size_t o = (size_t)(t0 + mm) * DIMC + nn;
      float v = acc[i] * g3[nn] * invs + b3[nn] + resid[o];
      dst[o] = fmaxf(v, 0.f);
    }
  }
}

// ---------------------------------------------------------------------------
// 12) final: out = LN(simple + complex)
// ---------------------------------------------------------------------------
__global__ void k_post(const float* __restrict__ s1, const float* __restrict__ s2,
                       const float* __restrict__ g, const float* __restrict__ b,
                       float* __restrict__ out) {
  int t0 = blockIdx.x * 16;
  int m = threadIdx.x >> 3, cg = (threadIdx.x & 7) * 16;
  size_t base = (size_t)(t0 + m) * DIMC;
  float v[16], s = 0.f, ss = 0.f;
#pragma unroll
  for (int j = 0; j < 16; ++j) {
    v[j] = s1[base + cg + j] + s2[base + cg + j];
    s += v[j]; ss += v[j] * v[j];
  }
  for (int off = 4; off; off >>= 1) { s += __shfl_xor(s, off, 32); ss += __shfl_xor(ss, off, 32); }
  float mean = s * (1.0f / DIMC);
  float var = ss * (1.0f / DIMC) - mean * mean;
  float inv = rsqrtf(var + EPSLN);
#pragma unroll
  for (int j = 0; j < 16; ++j)
    out[base + cg + j] = (v[j] - mean) * inv * g[cg + j] + b[cg + j];
}

// ---------------------------------------------------------------------------
// host-side launch
// ---------------------------------------------------------------------------
static constexpr size_t al256(size_t x) { return (x + 255) & ~(size_t)255; }

// workspace layout (bytes)
static constexpr size_t OFF_CE   = 0;
static constexpr size_t OFF_IMG  = al256(OFF_CE + (size_t)6 * 128 * 4);
static constexpr size_t OFF_KEY  = al256(OFF_IMG + (size_t)6144 * 128 * 4);
static constexpr size_t OFF_VAL  = al256(OFF_KEY + (size_t)6144 * 128 * 4);
static constexpr size_t OFF_QH1  = al256(OFF_VAL + (size_t)6144 * 128 * 4);
static constexpr size_t OFF_KH1  = al256(OFF_QH1 + (size_t)98304 * 128 * 2);
static constexpr size_t OFF_VH1  = al256(OFF_KH1 + (size_t)6144 * 128 * 2);
static constexpr size_t OFF_KH2  = al256(OFF_VH1 + (size_t)6144 * 128 * 2);
static constexpr size_t OFF_VH2  = al256(OFF_KH2 + (size_t)6144 * 128 * 2);
static constexpr size_t OFF_Q1   = al256(OFF_VH2 + (size_t)6144 * 128 * 2);
static constexpr size_t OFF_Q1B  = al256(OFF_Q1 + (size_t)16384 * 128 * 4);
static constexpr size_t OFF_QH2  = al256(OFF_Q1B + (size_t)16384 * 128 * 4);
static constexpr size_t OFF_A2   = al256(OFF_QH2 + (size_t)16384 * 128 * 2);
static constexpr size_t OFF_SIMP = al256(OFF_A2 + (size_t)16384 * 128 * 4);
static constexpr size_t OFF_Y1   = al256(OFF_SIMP + (size_t)16384 * 128 * 4);
static constexpr size_t OFF_Y2   = al256(OFF_Y1 + (size_t)16384 * 32 * 4);
static constexpr size_t OFF_BT1  = al256(OFF_Y2 + (size_t)16384 * 32 * 4);
static constexpr size_t OFF_CPLX = al256(OFF_BT1 + (size_t)16384 * 128 * 4);

// flattened param indices (setup_inputs insertion order)
enum {
  P_CAMW = 0, P_IMGW, P_BEVW, P_BEVB,
  P_FPG, P_FPB, P_FPW, P_FLG, P_FLB, P_FLW,
  A1_LNQG, A1_LNQB, A1_WQ, A1_BQ, A1_LNKG, A1_LNKB, A1_WK, A1_BK,
  A1_LNVG, A1_LNVB, A1_WV, A1_BV, A1_WP, A1_BP,
  A2_LNQG, A2_LNQB, A2_WQ, A2_BQ, A2_LNKG, A2_LNKB, A2_WK, A2_BK,
  A2_LNVG, A2_LNVB, A2_WV, A2_BV, A2_WP, A2_BP,
  P_PRE1G, P_PRE1B, M1_W1, M1_B1, M1_W2, M1_B2,
  P_PRE2G, P_PRE2B, M2_W1, M2_B1, M2_W2, M2_B2,
  B0_W1, B0_G1, B0_B1, B0_W2, B0_G2, B0_B2, B0_W3, B0_G3, B0_B3,
  B1_W1, B1_G1, B1_B1, B1_W2, B1_G2, B1_B2, B1_W3, B1_G3, B1_B3,
  P_POSTG, P_POSTB,
  IN_X, IN_FEAT, IN_IINV, IN_EINV, IN_INDEX
};

extern "C" void kernel_launch(void* const* d_in, const int* in_sizes, int n_in,
                              void* d_out, int out_size, void* d_ws, size_t ws_size,
                              hipStream_t stream) {
  (void)in_sizes; (void)n_in; (void)out_size; (void)ws_size;
  auto fin = [&](int i) { return (const float*)d_in[i]; };
  char* ws = (char*)d_ws;
  float*  ce   = (float*)(ws + OFF_CE);
  float*  imgT = (float*)(ws + OFF_IMG);
  float*  keyT = (float*)(ws + OFF_KEY);
  float*  valT = (float*)(ws + OFF_VAL);
  bf16_t* Qh1  = (bf16_t*)(ws + OFF_QH1);
  bf16_t* Kh1  = (bf16_t*)(ws + OFF_KH1);
  bf16_t* Vh1  = (bf16_t*)(ws + OFF_VH1);
  bf16_t* Kh2  = (bf16_t*)(ws + OFF_KH2);
  bf16_t* Vh2  = (bf16_t*)(ws + OFF_VH2);
  float*  q1   = (float*)(ws + OFF_Q1);
  float*  q1b  = (float*)(ws + OFF_Q1B);
  bf16_t* Qh2  = (bf16_t*)(ws + OFF_QH2);
  float*  a2   = (float*)(ws + OFF_A2);
  float*  simp = (float*)(ws + OFF_SIMP);
  float*  y1   = (float*)(ws + OFF_Y1);
  float*  y2   = (float*)(ws + OFF_Y2);
  float*  bt1  = (float*)(ws + OFF_BT1);
  float*  cplx = (float*)(ws + OFF_CPLX);

  const float* xin  = fin(IN_X);
  const float* feat = fin(IN_FEAT);

  k_cam_embed<<<6, 128, 0, stream>>>(fin(P_CAMW), fin(IN_EINV), ce);
  k_img_embed<<<6144, 128, 0, stream>>>(fin(P_IMGW), fin(IN_IINV), fin(IN_EINV), ce, imgT);
  k_keyval<<<384, 128, 0, stream>>>(feat, fin(P_FPG), fin(P_FPB), fin(P_FPW),
                                    fin(P_FLG), fin(P_FLB), fin(P_FLW), imgT, keyT, valT);
  k_build_q1<<<6144, 128, 0, stream>>>(xin, fin(P_BEVW), fin(P_BEVB), ce,
                                       fin(A1_LNQG), fin(A1_LNQB), fin(A1_WQ), fin(A1_BQ), Qh1);
  k_ln_linear<<<384, 128, 0, stream>>>(keyT, fin(A1_LNKG), fin(A1_LNKB), fin(A1_WK), fin(A1_BK), Kh1);
  k_ln_linear<<<384, 128, 0, stream>>>(valT, fin(A1_LNVG), fin(A1_LNVB), fin(A1_WV), fin(A1_BV), Vh1);
  k_ln_linear<<<384, 128, 0, stream>>>(keyT, fin(A2_LNKG), fin(A2_LNKB), fin(A2_WK), fin(A2_BK), Kh2);
  k_ln_linear<<<384, 128, 0, stream>>>(valT, fin(A2_LNVG), fin(A2_LNVB), fin(A2_WV), fin(A2_BV), Vh2);
  k_init_q1<<<16384, 128, 0, stream>>>(xin, q1);
  k_attn<0><<<dim3(6, 64), 128, 0, stream>>>(Qh1, Kh1, Vh1, fin(A1_WP), fin(A1_BP), nullptr, q1);
  k_mlp<<<1024, 128, 0, stream>>>(q1, fin(P_PRE1G), fin(P_PRE1B),
                                  fin(M1_W1), fin(M1_B1), fin(M1_W2), fin(M1_B2), q1b);
  k_ln_linear<<<1024, 128, 0, stream>>>(q1b, fin(A2_LNQG), fin(A2_LNQB), fin(A2_WQ), fin(A2_BQ), Qh2);
  k_attn<1><<<dim3(1, 64), 128, 0, stream>>>(Qh2, Kh2, Vh2, fin(A2_WP), fin(A2_BP), q1b, a2);
  k_mlp<<<1024, 128, 0, stream>>>(a2, fin(P_PRE2G), fin(P_PRE2B),
                                  fin(M2_W1), fin(M2_B1), fin(M2_W2), fin(M2_B2), simp);
  // conv expert (2x bottleneck) on q1b
  k_bneck_a<<<1024, 64, 0, stream>>>(q1b, fin(B0_W1), fin(B0_G1), fin(B0_B1), y1);
  k_conv3<<<64, 256, 0, stream>>>(y1, fin(B0_W2), fin(B0_G2), fin(B0_B2), y2);
  k_bneck_c<<<1024, 128, 0, stream>>>(y2, fin(B0_W3), fin(B0_G3), fin(B0_B3), q1b, bt1);
  k_bneck_a<<<1024, 64, 0, stream>>>(bt1, fin(B1_W1), fin(B1_G1), fin(B1_B1), y1);
  k_conv3<<<64, 256, 0, stream>>>(y1, fin(B1_W2), fin(B1_G2), fin(B1_B2), y2);
  k_bneck_c<<<1024, 128, 0, stream>>>(y2, fin(B1_W3), fin(B1_G3), fin(B1_B3), bt1, cplx);
  k_post<<<1024, 128, 0, stream>>>(simp, cplx, fin(P_POSTG), fin(P_POSTB), (float*)d_out);
}
